// SubgraphEncoder_24618752541031
// MI455X (gfx1250) — matrix operand order repaired and checked by reading, not compiled, not run on hardware
//
#include <hip/hip_runtime.h>
#include <hip/hip_bf16.h>

// ---------------- problem constants (match reference) ----------------
#define NNODES   100000
#define NEDGES   500000
#define NODE_IN  60
#define EDGE_IN  8
#define HDIM     256
#define NLAYERS  8
#define NGRAPH   64
#define BN_EPS   1e-5f

// ---------------- CDNA5 WMMA types ----------------
typedef __attribute__((ext_vector_type(16))) __bf16        v16bf;
typedef __attribute__((ext_vector_type(8)))  float         v8f;
typedef __attribute__((ext_vector_type(4)))  unsigned int  v4u;
typedef __attribute__((ext_vector_type(2)))  float         v2f;
typedef __attribute__((ext_vector_type(2)))  __bf16        v2bf;

union FragV { v16bf v; v4u q[2]; };

__device__ __forceinline__ unsigned short f2bf(float f) {
  // round-to-nearest-even fp32 -> bf16 (scalar epilogue use)
  unsigned int x = __float_as_uint(f);
  unsigned int r = x + 0x7FFFu + ((x >> 16) & 1u);
  return (unsigned short)(r >> 16);
}

#if __has_builtin(__builtin_amdgcn_cvt_pk_bf16_f32)
__device__ __forceinline__ unsigned int pack2(float lo, float hi) {
  union { v2bf v; unsigned int u; } t;
  t.v = __builtin_amdgcn_cvt_pk_bf16_f32(lo, hi);
  return t.u;
}
#else
__device__ __forceinline__ unsigned int pack2(float lo, float hi) {
  // let the backend select v_cvt_pk_bf16_f32 / v_cvt_bf16_f32
  v2f in; in.x = lo; in.y = hi;
  union { v2bf v; unsigned int u; } t;
  t.v = __builtin_convertvector(in, v2bf);
  return t.u;
}
#endif

// =====================================================================
// Fused GEMM:  out = epilogue( A' @ W + bias ),  W is [K,256] fp32.
// MODE 0 (KP=64,  K=60):  A' = A fp32,            epi = ReLU(BN), out fp32
// MODE 1 (KP=256, K=256): A' = (1+eps)*A + A2,    epi = ReLU,     out bf16
// MODE 2 (KP=256, K=256): A' = Abf (async-to-LDS) epi = ReLU(BN), out fp32
// Block: 256 threads = 8 wave32; 32(M) x 256(N) tile.
// LDS holds the full 32xKP A panel (bf16 pairs packed in dwords) and a
// per-K-step 32x256 B panel stored transposed+pair-packed so every WMMA
// fragment is two contiguous 16B LDS reads (ds_load_b128).
// =====================================================================
template<int MODE, int KP, int K, int LDA>
__global__ __launch_bounds__(256)
void gemm_wmma(const float* __restrict__ A,
               const float* __restrict__ A2,
               const unsigned short* __restrict__ Abf,
               const float* __restrict__ epsP,
               const float* __restrict__ W,
               const float* __restrict__ bias,
               const float* __restrict__ gamma,
               const float* __restrict__ beta,
               const float* __restrict__ mean,
               const float* __restrict__ var,
               float* __restrict__ outF,
               unsigned short* __restrict__ outBF)
{
  constexpr int PR = KP / 2;   // bf16 pairs (dwords) per A row
  constexpr int RS = PR + 4;   // padded A row stride in dwords (mult of 4)
  constexpr int BS = 20;       // padded B^T row stride in dwords (16 + 4)

  __shared__ __align__(16) unsigned int sA [32 * RS];
  __shared__ __align__(16) unsigned int sBt[256 * BS];

  const int tid  = threadIdx.x;
  const int wid  = tid >> 5;
  const int lane = tid & 31;
  const int g    = lane >> 4;    // lane half (ISA layout)
  const int ln   = lane & 15;
  const int rowbase = blockIdx.x * 32;
  const int mt  = wid & 1;       // M sub-tile
  const int nt0 = (wid >> 1) * 4;// first of 4 N tiles for this wave

  // ---------------- stage full A panel (32 x KP) ----------------
  if (MODE == 0 || MODE == 1) {
    float scale = 1.0f;
    if (MODE == 1) scale = 1.0f + epsP[0];
    #pragma unroll
    for (int i = 0; i < (32 * PR) / 256; ++i) {
      int s = tid + i * 256;
      int r = s / PR;
      int p = s % PR;
      float lo = 0.f, hi = 0.f;
      if (2 * p < K) {           // K even: pair fully in range (folds at CT)
        const float2 av = *((const float2*)(A + (size_t)(rowbase + r) * LDA) + p);
        lo = av.x; hi = av.y;
        if (MODE == 1) {
          const float2 gv = *((const float2*)(A2 + (size_t)(rowbase + r) * LDA) + p);
          lo = scale * lo + gv.x;
          hi = scale * hi + gv.y;
        }
      }
      sA[r * RS + p] = pack2(lo, hi);
    }
  } else {
    // MODE 2: rows are already bf16 row-major == packed-pair dword format.
    // Copy straight into LDS with the gfx1250 async-to-LDS engine.
    constexpr int SEGS = KP / 8;             // 16B segments per row
    const unsigned sA_base = (unsigned)(size_t)(&sA[0]);
    #pragma unroll
    for (int i = 0; i < (32 * SEGS) / 256; ++i) {
      int s = tid + i * 256;
      int r = s / SEGS;
      int seg = s % SEGS;
      unsigned lds_off = sA_base + (unsigned)(r * RS * 4 + seg * 16);
      const unsigned short* gp = Abf + (size_t)(rowbase + r) * HDIM + seg * 8;
      asm volatile("global_load_async_to_lds_b128 %0, %1, off"
                   :: "v"(lds_off), "v"(gp)
                   : "memory");
    }
    asm volatile("s_wait_asynccnt 0x0" ::: "memory");
  }

  v8f acc0 = {}, acc1 = {}, acc2 = {}, acc3 = {};

  #pragma unroll
  for (int k0 = 0; k0 < KP; k0 += 32) {
    __syncthreads();   // A visible (first iter); sBt safe to overwrite
    // ---- stage B^T pair-packed: sBt[c*BS + p] = {K=k0+2p, k0+2p+1} ----
    #pragma unroll
    for (int i = 0; i < 16; ++i) {
      int s = tid + i * 256;
      int c = s & 255;
      int p = s >> 8;            // 0..15 (uniform = i)
      int k = k0 + 2 * p;
      float lo = 0.f, hi = 0.f;
      if (k < K) {               // folds at compile time per unrolled step
        lo = W[(size_t)k * HDIM + c];
        hi = W[(size_t)(k + 1) * HDIM + c];
      }
      sBt[c * BS + p] = pack2(lo, hi);
    }
    __syncthreads();

    // ---- A fragment: two contiguous 16B reads ----
    FragV fa;
    {
      const unsigned int* ap = sA + (mt * 16 + ln) * RS + (k0 >> 1) + g * 4;
      fa.q[0] = *(const v4u*)(ap);
      fa.q[1] = *(const v4u*)(ap + 8);
    }
    // ---- 4 B fragments + WMMA ----
    #pragma unroll
    for (int t = 0; t < 4; ++t) {
      FragV fb;
      const unsigned int* bp = sBt + ((nt0 + t) * 16 + ln) * BS + g * 4;
      fb.q[0] = *(const v4u*)(bp);
      fb.q[1] = *(const v4u*)(bp + 8);
      v8f& acc = (t == 0) ? acc0 : (t == 1) ? acc1 : (t == 2) ? acc2 : acc3;
      acc = __builtin_amdgcn_wmma_f32_16x16x32_bf16(
          /*neg_a=*/false, fa.v, /*neg_b=*/false, fb.v,
          /*c_mod=*/(short)0, acc, /*reuse_a=*/false, /*reuse_b=*/false);
    }
  }

  // ---------------- epilogue ----------------
  #pragma unroll
  for (int t = 0; t < 4; ++t) {
    v8f acc = (t == 0) ? acc0 : (t == 1) ? acc1 : (t == 2) ? acc2 : acc3;
    int col = (nt0 + t) * 16 + ln;
    float b = bias[col];
    float bn_s = 0.0f, bn_o = 0.0f;
    if (MODE == 0 || MODE == 2) {
      bn_s = gamma[col] * rsqrtf(var[col] + BN_EPS);
      bn_o = beta[col] - mean[col] * bn_s;
    }
    #pragma unroll
    for (int v = 0; v < 8; ++v) {
      int row = rowbase + mt * 16 + g * 8 + v;   // C/D layout: M = g*8 + v
      float val = acc[v] + b;
      if (MODE == 0 || MODE == 2) val = val * bn_s + bn_o;
      val = fmaxf(val, 0.0f);
      if (MODE == 1) outBF[(size_t)row * HDIM + col] = f2bf(val);
      else           outF [(size_t)row * HDIM + col] = val;
    }
  }
}

// =====================================================================
// Fused edge stage: m = ReLU(h[src] + edge_attr @ We + be); agg[dst] += m
// One wave per edge; lane handles 8 coalesced columns.
// =====================================================================
__global__ __launch_bounds__(256)
void edge_message(const float* __restrict__ h,
                  const int* __restrict__ eidx,     // [2,E]
                  const float* __restrict__ eattr,  // [E,8]
                  const float* __restrict__ We,     // [8,256]
                  const float* __restrict__ be,     // [256]
                  float* __restrict__ agg)
{
  __shared__ float sW[EDGE_IN][HDIM];
  __shared__ float sb[HDIM];
  for (int i = threadIdx.x; i < EDGE_IN * HDIM; i += 256) sW[i >> 8][i & 255] = We[i];
  for (int i = threadIdx.x; i < HDIM; i += 256) sb[i] = be[i];
  __syncthreads();

  const int lane   = threadIdx.x & 31;
  const int waveId = blockIdx.x * 8 + (threadIdx.x >> 5);
  const int nWaves = gridDim.x * 8;

  for (int e = waveId; e < NEDGES; e += nWaves) {
    int src = eidx[e];
    int dst = eidx[NEDGES + e];
    float ea[EDGE_IN];
    #pragma unroll
    for (int i = 0; i < EDGE_IN; ++i) ea[i] = eattr[(size_t)e * EDGE_IN + i];
    const float* hs = h   + (size_t)src * HDIM;
    float*       ad = agg + (size_t)dst * HDIM;
    #pragma unroll
    for (int j = 0; j < 8; ++j) {
      int c = lane + 32 * j;
      float v = sb[c];
      #pragma unroll
      for (int i = 0; i < EDGE_IN; ++i) v += ea[i] * sW[i][c];
      v += hs[c];
      v = fmaxf(v, 0.0f);
      atomicAdd(&ad[c], v);
    }
  }
}

// =====================================================================
__global__ void zero_f4(float4* __restrict__ p, long n4) {
  long i = (long)blockIdx.x * blockDim.x + threadIdx.x;
  long stride = (long)gridDim.x * blockDim.x;
  float4 z = make_float4(0.f, 0.f, 0.f, 0.f);
  for (; i < n4; i += stride) p[i] = z;
}

// one wave per node: batch id loaded once, 8 coalesced column atomics/lane
__global__ __launch_bounds__(256)
void pool_accum(const float* __restrict__ h, const int* __restrict__ batch,
                float* __restrict__ gsum, float* __restrict__ cnt) {
  const int lane   = threadIdx.x & 31;
  const int waveId = blockIdx.x * 8 + (threadIdx.x >> 5);
  const int nWaves = gridDim.x * 8;
  for (int n = waveId; n < NNODES; n += nWaves) {
    int b = batch[n];
    const float* hn = h + (size_t)n * HDIM;
    float* gs = gsum + (size_t)b * HDIM;
    #pragma unroll
    for (int j = 0; j < 8; ++j) {
      int c = lane + 32 * j;
      atomicAdd(&gs[c], hn[c]);
    }
    if (lane == 0) atomicAdd(&cnt[b], 1.0f);
  }
}

__global__ __launch_bounds__(128)
void head_kernel(const float* __restrict__ gsum, const float* __restrict__ cnt,
                 const float* __restrict__ W1, const float* __restrict__ b1,
                 const float* __restrict__ W2, const float* __restrict__ b2,
                 float* __restrict__ out) {
  __shared__ float sg[HDIM];
  __shared__ float red[128];
  const int gph = blockIdx.x;
  const int j   = threadIdx.x;
  float c = fmaxf(cnt[gph], 1.0f);
  for (int i = j; i < HDIM; i += 128) sg[i] = gsum[(size_t)gph * HDIM + i] / c;
  __syncthreads();
  float hid = b1[j];
  for (int k = 0; k < HDIM; ++k) hid += sg[k] * W1[(size_t)k * (HDIM / 2) + j];
  hid = fmaxf(hid, 0.0f);
  red[j] = hid * W2[j];
  __syncthreads();
  for (int s = 64; s > 0; s >>= 1) {
    if (j < s) red[j] += red[j + s];
    __syncthreads();
  }
  if (j == 0) out[gph] = red[0] + b2[0];
}

// =====================================================================
extern "C" void kernel_launch(void* const* d_in, const int* in_sizes, int n_in,
                              void* d_out, int out_size, void* d_ws, size_t ws_size,
                              hipStream_t stream) {
  (void)in_sizes; (void)n_in; (void)out_size; (void)ws_size;
  const float* x     = (const float*)d_in[0];
  const int*   eidx  = (const int*)  d_in[1];
  const float* eattr = (const float*)d_in[2];
  const int*   batch = (const int*)  d_in[3];
  const float* xW    = (const float*)d_in[4];
  const float* xb    = (const float*)d_in[5];
  const float* xg    = (const float*)d_in[6];
  const float* xbe   = (const float*)d_in[7];
  const float* xm    = (const float*)d_in[8];
  const float* xv    = (const float*)d_in[9];
  const float* eW    = (const float*)d_in[10];
  const float* eb    = (const float*)d_in[11];
  const float* W1    = (const float*)d_in[12];
  const float* b1    = (const float*)d_in[13];
  const float* W2    = (const float*)d_in[14];
  const float* b2    = (const float*)d_in[15];
  const float* eps   = (const float*)d_in[16];
  const float* bng   = (const float*)d_in[17];
  const float* bnb   = (const float*)d_in[18];
  const float* bnm   = (const float*)d_in[19];
  const float* bnv   = (const float*)d_in[20];
  const float* hW1   = (const float*)d_in[21];
  const float* hb1   = (const float*)d_in[22];
  const float* hW2   = (const float*)d_in[23];
  const float* hb2   = (const float*)d_in[24];
  float* out = (float*)d_out;

  // -------- workspace layout --------
  char* ws = (char*)d_ws;
  const size_t szH = (size_t)NNODES * HDIM;                   // 25.6M elems
  float*          h    = (float*)(ws);                        // 102.4 MB
  float*          agg  = (float*)(ws + szH * 4);              // 102.4 MB
  unsigned short* tbf  = (unsigned short*)(ws + szH * 8);     //  51.2 MB
  float*          gsum = (float*)(ws + szH * 8 + szH * 2);    //  64 KB
  float*          cnt  = gsum + (size_t)NGRAPH * HDIM;        // 256 B

  dim3 blk(256);
  const int gemmGrid = NNODES / 32;  // 3125, exact

  // x_proj: Linear(60->256) + BN + ReLU  -> h
  gemm_wmma<0, 64, NODE_IN, NODE_IN><<<gemmGrid, blk, 0, stream>>>(
      x, nullptr, nullptr, nullptr, xW, xb, xg, xbe, xm, xv,
      h, nullptr);

  for (int l = 0; l < NLAYERS; ++l) {
    zero_f4<<<25000, blk, 0, stream>>>((float4*)agg, (long)(szH / 4));
    edge_message<<<2048, blk, 0, stream>>>(
        h, eidx, eattr, eW + (size_t)l * EDGE_IN * HDIM, eb + (size_t)l * HDIM, agg);
    // t = ReLU(((1+eps)h + agg) @ W1 + b1) -> bf16
    gemm_wmma<1, 256, HDIM, HDIM><<<gemmGrid, blk, 0, stream>>>(
        h, agg, nullptr, eps + l,
        W1 + (size_t)l * HDIM * HDIM, b1 + (size_t)l * HDIM,
        nullptr, nullptr, nullptr, nullptr,
        nullptr, tbf);
    // h = ReLU(BN(t @ W2 + b2))
    gemm_wmma<2, 256, HDIM, HDIM><<<gemmGrid, blk, 0, stream>>>(
        nullptr, nullptr, tbf, nullptr,
        W2 + (size_t)l * HDIM * HDIM, b2 + (size_t)l * HDIM,
        bng + (size_t)l * HDIM, bnb + (size_t)l * HDIM,
        bnm + (size_t)l * HDIM, bnv + (size_t)l * HDIM,
        h, nullptr);
  }

  // global mean pool + head
  zero_f4<<<16, blk, 0, stream>>>((float4*)gsum, (long)(NGRAPH * HDIM / 4));
  zero_f4<<<1, 64, 0, stream>>>((float4*)cnt, (long)(NGRAPH / 4));
  pool_accum<<<1024, blk, 0, stream>>>(h, batch, gsum, cnt);
  head_kernel<<<NGRAPH, 128, 0, stream>>>(gsum, cnt, hW1, hb1, hW2, hb2, out);
}
